// DifferentiableLengthRegulator_50697793962729
// MI455X (gfx1250) — compile-verified, matrix operand order checked by reading, not run on hardware
//
#include <hip/hip_runtime.h>

// ---------------------------------------------------------------------------
// DifferentiableLengthRegulator, fused for gfx1250 (MI455X).
//   B=32, C=256, T=512, L=2048
// f32 weight math + softmax sums -> f16 WMMA GEMM (v_wmma_f32_16x16x32_f16)
// with double-buffered async global->LDS staging (ASYNCcnt) -> f32 epilogue.
// ---------------------------------------------------------------------------

#define B_  32
#define C_  256
#define T_  512
#define L_  2048

#define NTILE 64           // L columns per workgroup
#define KTILE 64           // T columns per staged x tile

#define XS_STRIDE 72       // halves per x-tile row (64 + 8 pad) = 144B (16B-aligned rows)
#define WT_STRIDE 520      // halves per wT row (512 + 8 pad)    = 1040B (16B-aligned rows)

#define XS_BUF_BYTES (C_ * XS_STRIDE * 2)     // 36864 per buffer
#define WT_BYTES     (NTILE * WT_STRIDE * 2)  // 66560
#define CEN_BYTES    (T_ * 4)
#define WS_BYTES     (T_ * 4)
#define XM_BYTES     (T_ * 4)
#define SUM_BYTES    (4 * NTILE * 4)
#define INV_BYTES    (NTILE * 4)
#define SMEM_TOTAL   (2 * XS_BUF_BYTES + WT_BYTES + CEN_BYTES + WS_BYTES + XM_BYTES + SUM_BYTES + INV_BYTES)

typedef __attribute__((ext_vector_type(16))) _Float16 v16h;
typedef __attribute__((ext_vector_type(8)))  _Float16 v8h;
typedef __attribute__((ext_vector_type(8)))  float    v8f;
typedef __attribute__((__vector_size__(4 * sizeof(int)))) int v4i;

// pointer-to-int4 in global (AS1) / LDS (AS3) address spaces, as the async
// builtin's parameters require
typedef __attribute__((address_space(1))) v4i* gbuf_p;
typedef __attribute__((address_space(3))) v4i* lbuf_p;

// Async global->LDS support (guarded: host pass / older toolchains fall back)
#if defined(__has_builtin)
#if __has_builtin(__builtin_amdgcn_global_load_async_to_lds_b128)
#define HAVE_ASYNC 1
#endif
#endif
#ifndef HAVE_ASYNC
#define HAVE_ASYNC 0
#endif

__device__ __forceinline__ v16h cat8(v8h lo, v8h hi) {
    return __builtin_shufflevector(lo, hi, 0,1,2,3,4,5,6,7,8,9,10,11,12,13,14,15);
}

template <int N>
__device__ __forceinline__ void async_wait_le() {
#if HAVE_ASYNC
#if __has_builtin(__builtin_amdgcn_s_wait_asynccnt)
    __builtin_amdgcn_s_wait_asynccnt((unsigned short)N);
#else
    if constexpr (N == 0) { asm volatile("s_wait_asynccnt 0x0" ::: "memory"); }
    else                  { asm volatile("s_wait_asynccnt 0x8" ::: "memory"); }
#endif
#endif
}

// Issue one KTILE-wide x-tile stage: 8 x 16B chunks per thread (256 threads).
__device__ __forceinline__ void stage_issue(const _Float16* __restrict__ gbase, // xh[b][0][k0]
                                            _Float16* __restrict__ lbase,       // LDS buffer
                                            int tid) {
    const int rb = tid >> 3;            // 0..31
    const int cb = (tid & 7) * 8;       // half offset in row: 0..56 (16B chunks)
#pragma unroll
    for (int i = 0; i < 8; ++i) {
        const int row = rb + i * 32;    // 0..255
        const _Float16* g = gbase + (size_t)row * T_ + cb;
        _Float16* l = lbase + (size_t)row * XS_STRIDE + cb;
#if HAVE_ASYNC
        __builtin_amdgcn_global_load_async_to_lds_b128(
            (gbuf_p)(g), (lbuf_p)(l), 0, 0);
#else
        *(v8h*)l = *(const v8h*)g;
#endif
    }
}

// ---------------------------------------------------------------------------
// Kernel 0: xh[b,c,t] = f16( x[b,c,t] * x_mask[b,t] )
// ---------------------------------------------------------------------------
__global__ __launch_bounds__(256) void dlr_xh_kernel(const float* __restrict__ x,
                                                     const float* __restrict__ xmask,
                                                     _Float16* __restrict__ xh) {
    const size_t gid  = (size_t)blockIdx.x * 256 + threadIdx.x;
    const size_t base = gid * 16;
    const int b  = (int)(base / ((size_t)C_ * T_));
    const int t0 = (int)(base % T_);
    const float* xm = xmask + (size_t)b * T_ + t0;
    v16h hv;
#pragma unroll
    for (int i = 0; i < 16; i += 4) {
        const float4 v = *(const float4*)(x + base + i);
        const float4 m = *(const float4*)(xm + i);
        hv[i + 0] = (_Float16)(v.x * m.x);
        hv[i + 1] = (_Float16)(v.y * m.y);
        hv[i + 2] = (_Float16)(v.z * m.z);
        hv[i + 3] = (_Float16)(v.w * m.w);
    }
    *(v16h*)(xh + base) = hv;
}

// ---------------------------------------------------------------------------
// Kernel 1: center[b,t] = inclusive_cumsum(w)[b,t] - 0.5*w[b,t]
// ---------------------------------------------------------------------------
__global__ __launch_bounds__(T_) void dlr_center_kernel(const float* __restrict__ w,
                                                        float* __restrict__ center) {
    const int b = blockIdx.x;
    const int t = threadIdx.x;
    __shared__ float s[T_];
    const float wv = w[b * T_ + t];
    s[t] = wv;
    __syncthreads();
#pragma unroll
    for (int off = 1; off < T_; off <<= 1) {
        float add = (t >= off) ? s[t - off] : 0.0f;
        __syncthreads();
        s[t] += add;
        __syncthreads();
    }
    center[b * T_ + t] = s[t] - 0.5f * wv;
}

// ---------------------------------------------------------------------------
// Kernel 2: fused weights + softmax + WMMA GEMM
// grid = (L_/NTILE, B_), block = 256 threads (8 waves)
// ---------------------------------------------------------------------------
__global__ __launch_bounds__(256) void dlr_main_kernel(
    const _Float16* __restrict__ xh,    // [B,C,T] f16 (pre-masked)
    const float* __restrict__ w,        // [B,T]
    const float* __restrict__ xmask,    // [B,1,T]
    const float* __restrict__ ymask,    // [B,1,L]
    const float* __restrict__ sscale,   // [1]
    const float* __restrict__ center,   // [B,T]
    float* __restrict__ out)            // [B,C,L]
{
    extern __shared__ char smem[];
    _Float16* xs0   = (_Float16*)(smem);
    _Float16* xs1   = (_Float16*)(smem + XS_BUF_BYTES);
    _Float16* wT    = (_Float16*)(smem + 2 * XS_BUF_BYTES);
    char* p = smem + 2 * XS_BUF_BYTES + WT_BYTES;
    float* cen_s = (float*)(p);                 p += CEN_BYTES;
    float* w_s   = (float*)(p);                 p += WS_BYTES;
    float* xm_s  = (float*)(p);                 p += XM_BYTES;
    float* sums  = (float*)(p);                 p += SUM_BYTES;
    float* inv_s = (float*)(p);

    const int tid   = threadIdx.x;
    const int wave  = tid >> 5;
    const int lane  = tid & 31;
    const int laneN = lane & 15;
    const int laneH = lane >> 4;
    const int l0    = blockIdx.x * NTILE;
    const int b     = blockIdx.y;

    const float s_scale = sscale[0];
    const _Float16* xb = xh + (size_t)b * C_ * T_;

    // kick off async stage of the first x tile immediately
    stage_issue(xb, xs0, tid);

    // --- stage per-token scalars ---
    {
        const int t0 = tid, t1 = tid + 256;
        cen_s[t0] = center[b * T_ + t0];
        cen_s[t1] = center[b * T_ + t1];
        w_s[t0]   = w[b * T_ + t0];
        w_s[t1]   = w[b * T_ + t1];
        xm_s[t0]  = xmask[b * T_ + t0];
        xm_s[t1]  = xmask[b * T_ + t1];
    }
    __syncthreads();

    // --- weight phase: softmax numerators exp(masked exp(-0.5*sigma^2*s)),
    //     stored transposed as f16 (wT[l][t]); column sums in f32 ---
    {
        const int lcol = tid & (NTILE - 1);
        const int tg   = tid >> 6;             // 4 groups of 128 t's
        const float pos = (float)(l0 + lcol);
        float psum = 0.0f;
        _Float16* wrow = wT + (size_t)lcol * WT_STRIDE;
#pragma unroll 4
        for (int i = 0; i < T_ / 4; ++i) {
            const int t = tg * (T_ / 4) + i;
            float d = pos - cen_s[t];
            d = fminf(fmaxf(d, 1e-4f), 1e4f);
            const float sg = d * w_s[t];
            const float logit = (xm_s[t] == 0.0f) ? -1e9f
                               : __expf(-0.5f * sg * sg * s_scale);
            const float e = __expf(logit);
            psum += e;
            wrow[t] = (_Float16)e;
        }
        sums[tg * NTILE + lcol] = psum;
    }
    __syncthreads();
    if (tid < NTILE) {
        const float tot = sums[tid] + sums[NTILE + tid] + sums[2 * NTILE + tid] + sums[3 * NTILE + tid];
        const float ym  = ymask[(size_t)b * L_ + l0 + tid];
        inv_s[tid] = ym / tot;                 // softmax 1/sum fused with y_mask
    }

    // --- GEMM: out_tile[256 x 64] = X[256 x 512] * Wt[512 x 64], double-buffered ---
    v8f acc[2][4];
#pragma unroll
    for (int rt = 0; rt < 2; ++rt)
#pragma unroll
        for (int ct = 0; ct < 4; ++ct)
            acc[rt][ct] = (v8f){0.f,0.f,0.f,0.f,0.f,0.f,0.f,0.f};

    const int NKT = T_ / KTILE;                // 8
    for (int kt = 0; kt < NKT; ++kt) {
        _Float16* cur = (kt & 1) ? xs1 : xs0;
        _Float16* nxt = (kt & 1) ? xs0 : xs1;
        const bool more = (kt + 1) < NKT;
        if (more) stage_issue(xb + (kt + 1) * KTILE, nxt, tid);
        // in-order async completion: <=8 outstanding means current tile landed
        if (more) async_wait_le<8>(); else async_wait_le<0>();
        __syncthreads();

#pragma unroll
        for (int ki = 0; ki < KTILE / 32; ++ki) {
            const int kkx = ki * 32;           // within x tile
            const int kkw = kt * KTILE + kkx;  // absolute t for wT

            v16h afr[2];
#pragma unroll
            for (int rt = 0; rt < 2; ++rt) {
                const int m = (wave * 2 + rt) * 16 + laneN;
                const _Float16* base = &cur[(size_t)m * XS_STRIDE + kkx + 8 * laneH];
                afr[rt] = cat8(*(const v8h*)base, *(const v8h*)(base + 16));
            }
            v16h bfr[4];
#pragma unroll
            for (int ct = 0; ct < 4; ++ct) {
                const int n = ct * 16 + laneN;
                const _Float16* base = &wT[(size_t)n * WT_STRIDE + kkw + 8 * laneH];
                bfr[ct] = cat8(*(const v8h*)base, *(const v8h*)(base + 16));
            }
#pragma unroll
            for (int rt = 0; rt < 2; ++rt)
#pragma unroll
                for (int ct = 0; ct < 4; ++ct)
                    acc[rt][ct] = __builtin_amdgcn_wmma_f32_16x16x32_f16(
                        false, afr[rt], false, bfr[ct],
                        (short)0, acc[rt][ct], false, false);
        }
        __syncthreads();
    }

    // --- epilogue: normalize (softmax 1/sum * y_mask) in f32 and store ---
    float* outb = out + (size_t)b * C_ * L_;
#pragma unroll
    for (int rt = 0; rt < 2; ++rt) {
        const int baseRow = (wave * 2 + rt) * 16 + 8 * laneH;
#pragma unroll
        for (int ct = 0; ct < 4; ++ct) {
            const int lc  = ct * 16 + laneN;
            const float sc = inv_s[lc];
            const int col = l0 + lc;
#pragma unroll
            for (int v = 0; v < 8; ++v) {
                outb[(size_t)(baseRow + v) * L_ + col] = acc[rt][ct][v] * sc;
            }
        }
    }
}

// ---------------------------------------------------------------------------
extern "C" void kernel_launch(void* const* d_in, const int* in_sizes, int n_in,
                              void* d_out, int out_size, void* d_ws, size_t ws_size,
                              hipStream_t stream) {
    const float* x      = (const float*)d_in[0];   // [B,C,T]
    const float* w      = (const float*)d_in[1];   // [B,T]
    const float* xmask  = (const float*)d_in[2];   // [B,1,T]
    const float* ymask  = (const float*)d_in[3];   // [B,1,L]
    const float* sscale = (const float*)d_in[4];   // [1]
    float* out = (float*)d_out;                    // [B,C,L]

    // workspace: xh (f16, B*C*T) then center (f32, B*T)
    _Float16* xh  = (_Float16*)d_ws;
    float* center = (float*)((char*)d_ws + (size_t)B_ * C_ * T_ * 2);

    (void)in_sizes; (void)n_in; (void)out_size; (void)ws_size;

    (void)hipFuncSetAttribute((const void*)dlr_main_kernel,
                              hipFuncAttributeMaxDynamicSharedMemorySize, SMEM_TOTAL);

    dlr_xh_kernel<<<dim3((B_ * C_ * T_) / (256 * 16)), dim3(256), 0, stream>>>(x, xmask, xh);
    dlr_center_kernel<<<dim3(B_), dim3(T_), 0, stream>>>(w, center);

    dim3 grid(L_ / NTILE, B_);
    dlr_main_kernel<<<grid, dim3(256), SMEM_TOTAL, stream>>>(
        xh, w, xmask, ymask, sscale, center, out);
}
